// GammaPhaseAttention_87454124081926
// MI455X (gfx1250) — compile-verified, matrix operand order checked by reading, not run on hardware
//
#include <hip/hip_runtime.h>
#include <math.h>

// ---------------------------------------------------------------------------
// GammaPhaseAttention for MI455X (gfx1250, wave32, WMMA).
// Single pass: scores live in LDS, attention written to HBM exactly once
// (the 268MB attention write is the roofline floor at 23.3 TB/s).
// ---------------------------------------------------------------------------

typedef __bf16 bf16_t;
typedef __attribute__((ext_vector_type(16))) __bf16 v16bf;
typedef __attribute__((ext_vector_type(4)))  __bf16 v4bf;
typedef __attribute__((ext_vector_type(8)))  float  v8f;
typedef __attribute__((ext_vector_type(4)))  float  v4f;

#define B_SZ 16
#define NQ   2048
#define NK   2048
#define DH   128
#define QT   16                 // q rows per workgroup tile
#define S_STRIDE 20             // padded stride (floats) of S[n][r] panel
#define PBF_STRIDE (NK + 8)     // padded stride (bf16) of Pbf[r][n]
#define VCHUNK 256              // k rows of V staged per chunk
#define VT_STRIDE (VCHUNK + 8)  // padded stride (bf16) of Vt[d][k]

#define S_BYTES   (NK * S_STRIDE * 4)      // 163840
#define PBF_BYTES (QT * PBF_STRIDE * 2)    //  65792
#define VT_BYTES  (DH * VT_STRIDE * 2)     //  67584
#define SMEM_BYTES (S_BYTES + PBF_BYTES + VT_BYTES)  // 297216 <= 320KB/WG

// 16-bit A/B fragment gather (ISA 7.12.2, 16-bit A 16x32):
//   lanes 0-15 : halves 0-7 -> K 0-7,  halves 8-15 -> K 16-23
//   lanes 16-31: halves 0-7 -> K 8-15, halves 8-15 -> K 24-31
// => per lane: two contiguous 8-element runs at +hi*8 and +hi*8+16.
__device__ __forceinline__ v16bf load_frag_f32row(const float* rowp, int hi) {
  const float* p0 = rowp + hi * 8;
  const float* p1 = p0 + 16;
  v16bf f;
#pragma unroll
  for (int i = 0; i < 8; ++i) f[i] = (bf16_t)p0[i];
#pragma unroll
  for (int i = 0; i < 8; ++i) f[8 + i] = (bf16_t)p1[i];
  return f;
}

__device__ __forceinline__ v16bf load_frag_lds_bf16(const bf16_t* p, int hi) {
  const bf16_t* p0 = p + hi * 8;   // 16B-aligned run -> ds_load_b128
  const bf16_t* p1 = p0 + 16;
  v16bf f;
#pragma unroll
  for (int i = 0; i < 8; ++i) f[i] = p0[i];
#pragma unroll
  for (int i = 0; i < 8; ++i) f[8 + i] = p1[i];
  return f;
}

__global__ __launch_bounds__(256) void GammaPhaseAttention_87454124081926_kernel(
    const float* __restrict__ Q, const float* __restrict__ K,
    const float* __restrict__ V, const float* __restrict__ phases,
    float* __restrict__ out, float* __restrict__ attn) {
  extern __shared__ char smem[];
  float*  S   = (float*)smem;                          // S[n][r], n<2048, r<16
  bf16_t* Pbf = (bf16_t*)(smem + S_BYTES);             // Pbf[r][n]
  bf16_t* Vt  = (bf16_t*)(smem + S_BYTES + PBF_BYTES); // Vt[d][k] per chunk

  const int b    = blockIdx.y;
  const int q0   = blockIdx.x * QT;
  const int tid  = threadIdx.x;
  const int wave = tid >> 5;
  const int lane = tid & 31;
  const int lr   = lane & 15;   // row (A) / column (B,C) index within tile
  const int hi   = lane >> 4;   // lane-group: selects K sub-offsets / C rows

  const float scale = 0.08838834764831845f;   // 1/sqrt(128)

  // ---- Phase 1: S = gate * scale * (Q K^T), 16 q-rows x 2048 keys --------
  const float* qrow = Q + ((size_t)b * NQ + q0 + lr) * DH;
  v16bf afrag[4];
#pragma unroll
  for (int kk = 0; kk < 4; ++kk) afrag[kk] = load_frag_f32row(qrow + kk * 32, hi);

  for (int ns = 0; ns < 16; ++ns) {
    const int n = wave * 256 + ns * 16 + lr;      // this lane's key column
    const float* krow = K + ((size_t)b * NK + n) * DH;
    v8f acc = {0.f, 0.f, 0.f, 0.f, 0.f, 0.f, 0.f, 0.f};
#pragma unroll
    for (int kk = 0; kk < 4; ++kk) {
      v16bf bfrag = load_frag_f32row(krow + kk * 32, hi);
      acc = __builtin_amdgcn_wmma_f32_16x16x32_bf16(
          false, afrag[kk], false, bfrag, (short)0, acc, false, false);
    }
    // gamma-phase gate (CURRENT_PHASE = 0) folded with 1/sqrt(dh)
    float phi = phases[n];
    float pd  = fminf(phi, 6.283185307179586f - phi);
    float g   = __expf(-pd * pd * 0.8105694691387022f) * scale; // 8/pi^2
    // C layout: lane col N=lr; rows M = hi*8 + j  -> two b128 LDS stores
    float* sp = &S[n * S_STRIDE + hi * 8];
    v4f lo  = {acc[0] * g, acc[1] * g, acc[2] * g, acc[3] * g};
    v4f hi4 = {acc[4] * g, acc[5] * g, acc[6] * g, acc[7] * g};
    *(v4f*)sp       = lo;
    *(v4f*)(sp + 4) = hi4;
  }
  __syncthreads();

  // ---- Phase 2: row softmax; write fp32 attention once; stash bf16 P ----
  {
    const int r2  = tid >> 4;   // q row 0..15 (16 lanes per row, same wave)
    const int sub = tid & 15;
    float m = -3.402823466e38f;
    for (int n = sub; n < NK; n += 16) m = fmaxf(m, S[n * S_STRIDE + r2]);
#pragma unroll
    for (int o = 8; o > 0; o >>= 1) m = fmaxf(m, __shfl_xor(m, o, 32));
    // sum pass: overwrite panel with exp(s-m) (each element owned by 1 thread)
    float s = 0.f;
    for (int n = sub; n < NK; n += 16) {
      float e = __expf(S[n * S_STRIDE + r2] - m);
      s += e;
      S[n * S_STRIDE + r2] = e;
    }
#pragma unroll
    for (int o = 8; o > 0; o >>= 1) s += __shfl_xor(s, o, 32);
    const float inv = 1.0f / s;
    // write pass: 4 consecutive columns per lane -> b128 global stores
    // (16 lanes * 16B = 256B contiguous per row segment; HBM write is the
    //  roofline floor, so this stream must be full-width)
    float*  arow = attn + ((size_t)b * NQ + q0 + r2) * NK;
    bf16_t* prow = Pbf + r2 * PBF_STRIDE;
    for (int n0 = sub * 4; n0 < NK; n0 += 64) {
      float p0 = S[(n0 + 0) * S_STRIDE + r2] * inv;
      float p1 = S[(n0 + 1) * S_STRIDE + r2] * inv;
      float p2 = S[(n0 + 2) * S_STRIDE + r2] * inv;
      float p3 = S[(n0 + 3) * S_STRIDE + r2] * inv;
      v4f pv = {p0, p1, p2, p3};
      *(v4f*)(arow + n0) = pv;                       // global_store_b128
      v4bf pb = {(bf16_t)p0, (bf16_t)p1, (bf16_t)p2, (bf16_t)p3};
      *(v4bf*)(prow + n0) = pb;                      // ds_store_b64
    }
  }
  __syncthreads();

  // ---- Phase 3: O = P @ V; stage V chunks transposed (bf16) in LDS ------
  v8f oacc = {0.f, 0.f, 0.f, 0.f, 0.f, 0.f, 0.f, 0.f};
  const int d0 = wave * 16;       // 16 output columns per wave
  for (int c = 0; c < NK / VCHUNK; ++c) {
    __syncthreads();              // previous chunk fully consumed
    const float* vbase = V + ((size_t)b * NK + c * VCHUNK) * DH;
#pragma unroll 4
    for (int i = 0; i < (VCHUNK * DH) / 256; ++i) {
      int idx = tid + i * 256;
      int k = idx >> 7;           // 0..255
      int d = idx & (DH - 1);     // 0..127 (coalesced global reads)
      Vt[d * VT_STRIDE + k] = (bf16_t)vbase[(size_t)k * DH + d];
    }
    __syncthreads();
    // prefetch next chunk (global_prefetch_b8) so the next staging pass
    // hits WGP$/L2 while WMMA runs: 256 threads x 4 x 128B = 128KB
    if (c + 1 < NK / VCHUNK) {
      const float* vnext = vbase + (size_t)VCHUNK * DH;
#pragma unroll
      for (int i = 0; i < (VCHUNK * DH * 4) / (256 * 128); ++i)
        __builtin_prefetch(vnext + (size_t)(tid + i * 256) * 32, 0, 0);
    }
#pragma unroll
    for (int ks = 0; ks < VCHUNK / 32; ++ks) {
      v16bf pa = load_frag_lds_bf16(Pbf + lr * PBF_STRIDE + c * VCHUNK + ks * 32, hi);
      v16bf vb = load_frag_lds_bf16(Vt + (d0 + lr) * VT_STRIDE + ks * 32, hi);
      oacc = __builtin_amdgcn_wmma_f32_16x16x32_bf16(
          false, pa, false, vb, (short)0, oacc, false, false);
    }
  }
  // C layout: lane col d = d0+lr; rows m = hi*8 + j
  float* obase = out + ((size_t)b * NQ + q0 + hi * 8) * DH + (d0 + lr);
#pragma unroll
  for (int j = 0; j < 8; ++j) obase[(size_t)j * DH] = oacc[j];
}

extern "C" void kernel_launch(void* const* d_in, const int* in_sizes, int n_in,
                              void* d_out, int out_size, void* d_ws, size_t ws_size,
                              hipStream_t stream) {
  (void)in_sizes; (void)n_in; (void)out_size; (void)d_ws; (void)ws_size;
  const float* Q      = (const float*)d_in[0];
  const float* K      = (const float*)d_in[1];
  const float* V      = (const float*)d_in[2];
  const float* phases = (const float*)d_in[3];
  float* out  = (float*)d_out;                       // [16,2048,128]
  float* attn = out + (size_t)B_SZ * NQ * DH;        // [16,2048,2048]

  // Allow >64KB dynamic LDS (CDNA5 WGP: up to 320KB per workgroup).
  (void)hipFuncSetAttribute(
      (const void*)GammaPhaseAttention_87454124081926_kernel,
      hipFuncAttributeMaxDynamicSharedMemorySize, SMEM_BYTES);

  dim3 grid(NQ / QT, B_SZ);   // 128 x 16 workgroups
  GammaPhaseAttention_87454124081926_kernel<<<grid, 256, SMEM_BYTES, stream>>>(
      Q, K, V, phases, out, attn);
}
